// re_feats_3917010174495
// MI455X (gfx1250) — matrix operand order repaired and checked by reading, not compiled, not run on hardware
//
#include <hip/hip_runtime.h>
#include <hip/hip_bf16.h>

// ---------------------------------------------------------------------------
// re_feats: s = leaky_relu(im @ aud^T); row/col l2norm+softmax; two recon GEMMs.
// All three 8192x8192x1024 GEMMs run on v_wmma_f32_16x16x32_bf16 (CDNA5).
// ---------------------------------------------------------------------------

typedef __attribute__((ext_vector_type(8)))  __bf16 v8bf;
typedef __attribute__((ext_vector_type(16))) __bf16 v16bf;
typedef __attribute__((ext_vector_type(8)))  float  v8f;

#define N_IM   8192
#define N_AUD  8192
#define DDIM   1024
#define TILE   128          // C-tile is 128x128, 8 waves (wave32) per block
#define KSTEP  32           // bf16 WMMA K
#define LDSS   40           // LDS row stride in bf16 elems (80B: 16B aligned, staggers banks)
#define SMOOTHF 6.0f
#define EPSF    1e-8f
#define NSLOPE  0.1f

// ---- WMMA fragment loads (ISA 7.12.2 layouts) ------------------------------
// A (16x32 bf16): lane l -> row M = base + (l&15); lane halves split K:
//   lanes 0-15 hold K=0..15; lanes 16-31 hold K=16..31 (contiguous, matches B)
__device__ __forceinline__ v16bf load_a_frag(const __bf16* ldsA, int mbase, int lane) {
    const __bf16* row = ldsA + (mbase + (lane & 15)) * LDSS;
    int h8 = (lane >> 4) * 16;
    v8bf lo = *(const v8bf*)(row + h8);
    v8bf hi = *(const v8bf*)(row + 8 + h8);
    return __builtin_shufflevector(lo, hi, 0,1,2,3,4,5,6,7,8,9,10,11,12,13,14,15);
}
// B (32x16 bf16): lane l -> col N = base + (l&15); lanes 0-15 hold K=0..15,
// lanes 16-31 hold K=16..31 (contiguous per lane, as in sparse-B tables).
__device__ __forceinline__ v16bf load_b_frag(const __bf16* ldsB, int nbase, int lane) {
    const __bf16* row = ldsB + (nbase + (lane & 15)) * LDSS + (lane >> 4) * 16;
    v8bf lo = *(const v8bf*)(row);
    v8bf hi = *(const v8bf*)(row + 8);
    return __builtin_shufflevector(lo, hi, 0,1,2,3,4,5,6,7,8,9,10,11,12,13,14,15);
}

// One K=32 step: wave computes its 64x32 sub-tile = 4x2 WMMAs.
__device__ __forceinline__ void mma_step(const __bf16* ldsA, const __bf16* ldsB,
                                         int lane, int wrow, int wcol, v8f (&acc)[4][2]) {
    v16bf b0 = load_b_frag(ldsB, wcol * 32,      lane);
    v16bf b1 = load_b_frag(ldsB, wcol * 32 + 16, lane);
#pragma unroll
    for (int mi = 0; mi < 4; ++mi) {
        v16bf a = load_a_frag(ldsA, wrow * 64 + mi * 16, lane);
        acc[mi][0] = __builtin_amdgcn_wmma_f32_16x16x32_bf16(
            false, a, false, b0, (short)0, acc[mi][0], false, false);
        acc[mi][1] = __builtin_amdgcn_wmma_f32_16x16x32_bf16(
            false, a, false, b1, (short)0, acc[mi][1], false, false);
    }
}

// ---- K1: f32 -> bf16 convert + transposed bf16 copy ------------------------
__global__ void cvt_tr_kernel(const float* __restrict__ src,
                              __bf16* __restrict__ dst, __bf16* __restrict__ dstT,
                              int R, int C) {
    __shared__ float tile[32][33];
    int c0 = blockIdx.x * 32, r0 = blockIdx.y * 32;
    int tx = threadIdx.x, ty = threadIdx.y;            // 32 x 8
#pragma unroll
    for (int i = 0; i < 32; i += 8) {
        float v = src[(size_t)(r0 + ty + i) * C + c0 + tx];
        tile[ty + i][tx] = v;
        dst[(size_t)(r0 + ty + i) * C + c0 + tx] = (__bf16)v;
    }
    __syncthreads();
#pragma unroll
    for (int i = 0; i < 32; i += 8)
        dstT[(size_t)(c0 + ty + i) * R + r0 + tx] = (__bf16)tile[tx][ty + i];
}

__global__ void zero_kernel(float* __restrict__ p, int n) {
    int i = blockIdx.x * 256 + threadIdx.x;
    if (i < n) p[i] = 0.0f;
}

// ---- K2: s = leaky_relu(im @ aud^T) in bf16, plus row/col sum-of-squares ---
__global__ __launch_bounds__(256)
void gemm_scores_kernel(const __bf16* __restrict__ imb, const __bf16* __restrict__ audb,
                        __bf16* __restrict__ sbf,
                        float* __restrict__ rowss, float* __restrict__ colss) {
    __shared__ __bf16 ldsA[TILE * LDSS];
    __shared__ __bf16 ldsB[TILE * LDSS];
    __shared__ float  outT[TILE][TILE + 1];

    int tid  = threadIdx.x;
    int lane = tid & 31, wave = tid >> 5;
    int wrow = wave >> 2, wcol = wave & 3;
    int i0 = blockIdx.y * TILE, j0 = blockIdx.x * TILE;

    v8f acc[4][2] = {};

    int r = tid >> 1, h = tid & 1;                      // 128 rows x 2 half-chunks
    const __bf16* srcA = imb  + (size_t)(i0 + r) * DDIM + h * 16;
    const __bf16* srcB = audb + (size_t)(j0 + r) * DDIM + h * 16;
    __bf16* dA = ldsA + r * LDSS + h * 16;
    __bf16* dB = ldsB + r * LDSS + h * 16;

    for (int k0 = 0; k0 < DDIM; k0 += KSTEP) {
        __syncthreads();
        *(uint4*)(dA)     = *(const uint4*)(srcA);
        *(uint4*)(dA + 8) = *(const uint4*)(srcA + 8);
        *(uint4*)(dB)     = *(const uint4*)(srcB);
        *(uint4*)(dB + 8) = *(const uint4*)(srcB + 8);
        srcA += KSTEP; srcB += KSTEP;
        __builtin_prefetch(srcA, 0, 0);                 // global_prefetch_b8
        __builtin_prefetch(srcB, 0, 0);
        __syncthreads();
        mma_step(ldsA, ldsB, lane, wrow, wcol, acc);
    }

    __syncthreads();                                    // done with ldsA/ldsB
#pragma unroll
    for (int mi = 0; mi < 4; ++mi)
#pragma unroll
        for (int ni = 0; ni < 2; ++ni) {
            float* f = (float*)&acc[mi][ni];
            int col = wcol * 32 + ni * 16 + (lane & 15);
#pragma unroll
            for (int r8 = 0; r8 < 8; ++r8) {
                int row = wrow * 64 + mi * 16 + r8 + 8 * (lane >> 4);
                float v = f[r8];
                outT[row][col] = v > 0.0f ? v : NSLOPE * v;   // leaky relu
            }
        }
    __syncthreads();

    if (tid < 128) {                                    // per-row sum of squares
        float s = 0.0f;
        for (int c = 0; c < TILE; ++c) { float v = outT[tid][c]; s += v * v; }
        atomicAdd(&rowss[i0 + tid], s);
    } else {                                            // per-col sum of squares
        int col = tid - 128; float s = 0.0f;
        for (int rr = 0; rr < TILE; ++rr) { float v = outT[rr][col]; s += v * v; }
        atomicAdd(&colss[j0 + col], s);
    }

    for (int rep = 0; rep < (TILE * TILE) / 256; ++rep) {   // store s tile bf16
        int idx = rep * 256 + tid;
        int row = idx >> 7, col = idx & 127;
        sbf[(size_t)(i0 + row) * N_AUD + j0 + col] = (__bf16)outT[row][col];
    }
}

// ---- K3: scales + softmax denominators -------------------------------------
__global__ void make_scales_kernel(const float* __restrict__ ss, float* __restrict__ sc, int n) {
    int i = blockIdx.x * 256 + threadIdx.x;
    if (i < n) sc[i] = SMOOTHF / (sqrtf(ss[i]) + EPSF);
}

__global__ __launch_bounds__(256)
void row_denom_kernel(const __bf16* __restrict__ sbf, const float* __restrict__ rowsc,
                      float* __restrict__ rowde) {
    int i = blockIdx.x;
    float a = rowsc[i];
    const __bf16* row = sbf + (size_t)i * N_AUD;
    float sum = 0.0f;
    for (int j = threadIdx.x; j < N_AUD; j += 256) sum += __expf(a * (float)row[j]);
    __shared__ float red[256];
    red[threadIdx.x] = sum;
    __syncthreads();
    for (int s = 128; s > 0; s >>= 1) {
        if (threadIdx.x < s) red[threadIdx.x] += red[threadIdx.x + s];
        __syncthreads();
    }
    if (threadIdx.x == 0) rowde[i] = red[0];
}

__global__ __launch_bounds__(256)
void col_denom_kernel(const __bf16* __restrict__ sbf, const float* __restrict__ colsc,
                      float* __restrict__ colde) {
    int j = blockIdx.x * 64 + (threadIdx.x & 63);
    int g = threadIdx.x >> 6;                           // 4 row-groups per column
    float a = colsc[j];
    float sum = 0.0f;
    for (int i = g; i < N_IM; i += 4)
        sum += __expf(a * (float)sbf[(size_t)i * N_AUD + j]);
    __shared__ float red[256];
    red[threadIdx.x] = sum;
    __syncthreads();
    if (g == 0)
        colde[j] = red[threadIdx.x] + red[threadIdx.x + 64] +
                   red[threadIdx.x + 128] + red[threadIdx.x + 192];
}

// ---- K4: im_rec = softmax-row(s) @ aud  (exp folded into A staging) --------
__global__ __launch_bounds__(256)
void gemm_imrec_kernel(const __bf16* __restrict__ sbf, const __bf16* __restrict__ audtb,
                       const float* __restrict__ rowsc, const float* __restrict__ rowde,
                       float* __restrict__ out) {
    __shared__ __bf16 ldsA[TILE * LDSS];
    __shared__ __bf16 ldsB[TILE * LDSS];
    __shared__ float  sScale[TILE];

    int tid = threadIdx.x, lane = tid & 31, wave = tid >> 5;
    int wrow = wave >> 2, wcol = wave & 3;
    int i0 = blockIdx.y * TILE;                         // im_rec rows
    int d0 = blockIdx.x * TILE;                         // feature cols

    if (tid < TILE) sScale[tid] = rowsc[i0 + tid];
    v8f acc[4][2] = {};

    int r = tid >> 1, h = tid & 1;
    const __bf16* srcA = sbf   + (size_t)(i0 + r) * N_AUD + h * 16;
    const __bf16* srcB = audtb + (size_t)(d0 + r) * N_AUD + h * 16;
    __bf16* dA = ldsA + r * LDSS + h * 16;
    __bf16* dB = ldsB + r * LDSS + h * 16;
    __syncthreads();
    float asc = sScale[r];

    for (int k0 = 0; k0 < N_AUD; k0 += KSTEP) {
        __syncthreads();
        v8bf x0 = *(const v8bf*)srcA;
        v8bf x1 = *(const v8bf*)(srcA + 8);
#pragma unroll
        for (int e = 0; e < 8; ++e) {
            dA[e]     = (__bf16)__expf(asc * (float)x0[e]);   // |arg|<=6, safe
            dA[8 + e] = (__bf16)__expf(asc * (float)x1[e]);
        }
        *(uint4*)(dB)     = *(const uint4*)(srcB);
        *(uint4*)(dB + 8) = *(const uint4*)(srcB + 8);
        srcA += KSTEP; srcB += KSTEP;
        __builtin_prefetch(srcA, 0, 0);
        __builtin_prefetch(srcB, 0, 0);
        __syncthreads();
        mma_step(ldsA, ldsB, lane, wrow, wcol, acc);
    }

#pragma unroll
    for (int mi = 0; mi < 4; ++mi)
#pragma unroll
        for (int ni = 0; ni < 2; ++ni) {
            float* f = (float*)&acc[mi][ni];
            int col = d0 + wcol * 32 + ni * 16 + (lane & 15);
#pragma unroll
            for (int r8 = 0; r8 < 8; ++r8) {
                int row = i0 + wrow * 64 + mi * 16 + r8 + 8 * (lane >> 4);
                out[(size_t)row * DDIM + col] = f[r8] / rowde[row];
            }
        }
}

// ---- K5: aud_rec = softmax-col(s)^T @ im (A staged transposed from s) ------
__global__ __launch_bounds__(256)
void gemm_audrec_kernel(const __bf16* __restrict__ sbf, const __bf16* __restrict__ imtb,
                        const float* __restrict__ colsc, const float* __restrict__ colde,
                        float* __restrict__ out) {
    __shared__ __bf16 ldsA[TILE * LDSS];
    __shared__ __bf16 ldsB[TILE * LDSS];
    __shared__ float  sScale[TILE];

    int tid = threadIdx.x, lane = tid & 31, wave = tid >> 5;
    int wrow = wave >> 2, wcol = wave & 3;
    int j0 = blockIdx.y * TILE;                         // aud_rec rows (cols of s)
    int d0 = blockIdx.x * TILE;

    if (tid < TILE) sScale[tid] = colsc[j0 + tid];
    v8f acc[4][2] = {};

    int r = tid >> 1, h = tid & 1;
    const __bf16* srcB = imtb + (size_t)(d0 + r) * N_IM + h * 16;
    __bf16* dB = ldsB + r * LDSS + h * 16;

    int jj  = tid & 127;                                // A transpose staging coords
    int kk2 = tid >> 7;                                 // 0/1
    __syncthreads();
    float csc = sScale[jj];

    for (int k0 = 0; k0 < N_IM; k0 += KSTEP) {
        __syncthreads();
#pragma unroll
        for (int it = 0; it < 16; ++it) {               // 32 k-rows, 2 per pass
            int kk = it * 2 + kk2;
            float v = (float)sbf[(size_t)(k0 + kk) * N_AUD + j0 + jj];
            ldsA[jj * LDSS + kk] = (__bf16)__expf(csc * v);
        }
        *(uint4*)(dB)     = *(const uint4*)(srcB);
        *(uint4*)(dB + 8) = *(const uint4*)(srcB + 8);
        srcB += KSTEP;
        __builtin_prefetch(srcB, 0, 0);
        __syncthreads();
        mma_step(ldsA, ldsB, lane, wrow, wcol, acc);
    }

#pragma unroll
    for (int mi = 0; mi < 4; ++mi)
#pragma unroll
        for (int ni = 0; ni < 2; ++ni) {
            float* f = (float*)&acc[mi][ni];
            int col = d0 + wcol * 32 + ni * 16 + (lane & 15);
#pragma unroll
            for (int r8 = 0; r8 < 8; ++r8) {
                int row = j0 + wrow * 64 + mi * 16 + r8 + 8 * (lane >> 4);
                out[(size_t)row * DDIM + col] = f[r8] / colde[row];
            }
        }
}

// ---------------------------------------------------------------------------
extern "C" void kernel_launch(void* const* d_in, const int* in_sizes, int n_in,
                              void* d_out, int out_size, void* d_ws, size_t ws_size,
                              hipStream_t stream) {
    const float* im  = (const float*)d_in[0];
    const float* aud = (const float*)d_in[1];
    float* out = (float*)d_out;

    // Workspace carve-up (~192.2 MB total)
    __bf16* imb   = (__bf16*)d_ws;                               // [8192,1024]
    __bf16* audb  = imb   + (size_t)N_IM  * DDIM;                // [8192,1024]
    __bf16* imtb  = audb  + (size_t)N_AUD * DDIM;                // [1024,8192]
    __bf16* audtb = imtb  + (size_t)DDIM  * N_IM;                // [1024,8192]
    __bf16* sbf   = audtb + (size_t)DDIM  * N_AUD;               // [8192,8192]
    float*  rowss = (float*)(sbf + (size_t)N_IM * N_AUD);
    float*  colss = rowss + N_IM;
    float*  rowde = colss + N_AUD;
    float*  colde = rowde + N_IM;
    float*  rowsc = colde + N_AUD;
    float*  colsc = rowsc + N_IM;

    dim3 cvtBlk(32, 8), cvtGrd(DDIM / 32, N_IM / 32);
    cvt_tr_kernel<<<cvtGrd, cvtBlk, 0, stream>>>(im,  imb,  imtb,  N_IM,  DDIM);
    cvt_tr_kernel<<<cvtGrd, cvtBlk, 0, stream>>>(aud, audb, audtb, N_AUD, DDIM);

    zero_kernel<<<(N_IM + N_AUD + 255) / 256, 256, 0, stream>>>(rowss, N_IM + N_AUD);

    gemm_scores_kernel<<<dim3(N_AUD / TILE, N_IM / TILE), 256, 0, stream>>>(
        imb, audb, sbf, rowss, colss);

    make_scales_kernel<<<N_IM  / 256, 256, 0, stream>>>(rowss, rowsc, N_IM);
    make_scales_kernel<<<N_AUD / 256, 256, 0, stream>>>(colss, colsc, N_AUD);

    row_denom_kernel<<<N_IM, 256, 0, stream>>>(sbf, rowsc, rowde);
    col_denom_kernel<<<N_AUD / 64, 256, 0, stream>>>(sbf, colsc, colde);

    gemm_imrec_kernel<<<dim3(DDIM / TILE, N_IM / TILE), 256, 0, stream>>>(
        sbf, audtb, rowsc, rowde, out);
    gemm_audrec_kernel<<<dim3(DDIM / TILE, N_AUD / TILE), 256, 0, stream>>>(
        sbf, imtb, colsc, colde, out + (size_t)N_IM * DDIM);
}
